// SambaYDecoderLayer_89043261981034
// MI455X (gfx1250) — compile-verified, matrix operand order checked by reading, not run on hardware
//
#include <hip/hip_runtime.h>
#include <hip/hip_bf16.h>
#include <math.h>

// ---- problem constants (match reference) ----
#define DMf   2048      // model dim
#define DIf   4096      // inner dim = 2*DM
#define LLf   2048      // sequence length
#define BBf   2         // batch
#define NSTf  16        // SSM state size
#define KCf   4         // conv kernel
#define RRf   128       // dt rank
#define FFDf  8192      // FF dim
#define MROWS (BBf*LLf) // 4096 rows for all GEMMs
#define LN_EPS 1e-5f

typedef __attribute__((ext_vector_type(16))) __bf16 v16bf;
typedef __attribute__((ext_vector_type(8)))  __bf16 v8bf;
typedef __attribute__((ext_vector_type(4)))  __bf16 v4bf;
typedef __attribute__((ext_vector_type(8)))  float  v8f;
typedef __attribute__((address_space(3)))    __bf16 lds_bf16;

__device__ __forceinline__ float siluf(float x)     { return x / (1.f + __expf(-x)); }
__device__ __forceinline__ float softplusf(float x) { return x > 20.f ? x : log1pf(__expf(x)); }

// CDNA5 async copy: 16B global -> LDS, tracked by ASYNCcnt, no VGPR round-trip.
__device__ __forceinline__ void async_cp16(lds_bf16* dst, const __bf16* src) {
    asm volatile("global_load_async_to_lds_b128 %0, %1, off"
                 :: "v"(dst), "v"(src) : "memory");
}
template<int N>
__device__ __forceinline__ void wait_asynccnt() {
    asm volatile("s_wait_asynccnt %0" :: "n"(N) : "memory");
}

// ----------------------------------------------------------------------------
// fp32 -> bf16 cast (contiguous, 8 elems/thread)
// ----------------------------------------------------------------------------
__global__ __launch_bounds__(256)
void cast_bf16_kernel(const float* __restrict__ src, __bf16* __restrict__ dst)
{
    const size_t i = ((size_t)blockIdx.x * 256 + threadIdx.x) * 8;
    const float4 a = *(const float4*)(src + i);
    const float4 b = *(const float4*)(src + i + 4);
    v8bf o;
    o[0]=(__bf16)a.x; o[1]=(__bf16)a.y; o[2]=(__bf16)a.z; o[3]=(__bf16)a.w;
    o[4]=(__bf16)b.x; o[5]=(__bf16)b.y; o[6]=(__bf16)b.z; o[7]=(__bf16)b.w;
    *(v8bf*)(dst + i) = o;
}

// strided cast: dt_bf[4096,128] <- ssm[4096,160][:, 0:128]
__global__ __launch_bounds__(256)
void cast_dt_kernel(const float* __restrict__ ssm, __bf16* __restrict__ dt)
{
    const int idx = blockIdx.x * 256 + threadIdx.x;   // over 4096*128/4
    const int row = idx >> 5;
    const int col = (idx & 31) << 2;
    const float4 f = *(const float4*)(ssm + (size_t)row*160 + col);
    v4bf o; o[0]=(__bf16)f.x; o[1]=(__bf16)f.y; o[2]=(__bf16)f.z; o[3]=(__bf16)f.w;
    *(v4bf*)(dt + (size_t)row*RRf + col) = o;
}

// ----------------------------------------------------------------------------
// LayerNorm: one block per row of DM=2048; writes bf16 (feeds a GEMM)
// ----------------------------------------------------------------------------
__global__ __launch_bounds__(256)
void ln_kernel(const float* __restrict__ in, const float* __restrict__ w,
               const float* __restrict__ b, __bf16* __restrict__ out)
{
    const int row = blockIdx.x;
    const int t   = threadIdx.x;
    const float* x = in  + (size_t)row * DMf;
    __bf16*      o = out + (size_t)row * DMf;

    float v[8];
    float s = 0.f;
#pragma unroll
    for (int i = 0; i < 8; ++i) { v[i] = x[t + i*256]; s += v[i]; }

    __shared__ float red[256];
    red[t] = s; __syncthreads();
    for (int off = 128; off > 0; off >>= 1) { if (t < off) red[t] += red[t+off]; __syncthreads(); }
    const float mu = red[0] * (1.f / DMf);
    __syncthreads();

    float s2 = 0.f;
#pragma unroll
    for (int i = 0; i < 8; ++i) { float d = v[i] - mu; s2 += d*d; }
    red[t] = s2; __syncthreads();
    for (int off = 128; off > 0; off >>= 1) { if (t < off) red[t] += red[t+off]; __syncthreads(); }
    const float rstd = rsqrtf(red[0] * (1.f / DMf) + LN_EPS);

#pragma unroll
    for (int i = 0; i < 8; ++i) {
        int c = t + i*256;
        o[c] = (__bf16)((v[i] - mu) * rstd * w[c] + b[c]);
    }
}

// ----------------------------------------------------------------------------
// Depthwise causal conv (K=4) + SiLU -> xs (fp32 for scan) and xs_bf (for GEMM)
// ----------------------------------------------------------------------------
__global__ __launch_bounds__(256)
void conv_silu_kernel(const float* __restrict__ xz, const float* __restrict__ cw,
                      const float* __restrict__ cb, float* __restrict__ xs,
                      __bf16* __restrict__ xsb)
{
    const size_t idx = (size_t)blockIdx.x * 256 + threadIdx.x;  // over B*L*DI
    const int    d   = (int)(idx % DIf);
    const size_t bl  = idx / DIf;
    const int    l   = (int)(bl % LLf);
    const size_t bb  = bl / LLf;

    float acc = cb[d];
#pragma unroll
    for (int k = 0; k < KCf; ++k) {
        int ls = l + k - (KCf - 1);
        if (ls >= 0)
            acc += xz[(bb*LLf + (size_t)ls) * (size_t)(2*DIf) + d] * cw[d*KCf + k];
    }
    const float r = siluf(acc);
    xs[idx]  = r;
    xsb[idx] = (__bf16)r;
}

// ----------------------------------------------------------------------------
// bf16 WMMA GEMM with async-LDS double-buffered staging.
//  C[M,Nout] = act( A[M,Kdim]_bf16 * W[Nout,Kdim]_bf16^T + bias ) + resid
//  256 threads = 8 waves; BM=128 BN=128 BK=32; wave tile 64x32 = 4x2 frags.
//  Every wave issues exactly 4 async copies per tile (OOB rows go to a dump
//  slot with a clamped source), so s_wait_asynccnt 4 == "current tile landed".
// ----------------------------------------------------------------------------
__global__ __launch_bounds__(256)
void gemm_wmma_bf16(const __bf16* __restrict__ A, int lda,
                    const __bf16* __restrict__ W, int Nout, int Kdim,
                    float* __restrict__ C, int ldc,
                    const float* __restrict__ bias,
                    const float* __restrict__ resid,
                    int actmode)
{
    constexpr int BM = 128, BN = 128, BK = 32, PK = BK + 8;
    __shared__ __align__(16) __bf16 As[2][BM*PK];
    __shared__ __align__(16) __bf16 Bs[2][BN*PK];
    __shared__ __align__(16) __bf16 dump[8];          // OOB async sink (16B)

    const int t    = threadIdx.x;
    const int lane = t & 31, wave = t >> 5;
    const int wm   = wave >> 2, wn = wave & 3;        // 2 x 4 wave grid
    const int lr   = lane & 15, lh = lane >> 4;
    const int m0   = blockIdx.y * BM;
    const int n0   = blockIdx.x * BN;

    // zero-fill LDS rows for out-of-range W rows (partial N tile, e.g. Nout=160)
    if (n0 + BN > Nout) {
#pragma unroll
        for (int i = 0; i < 2; ++i) {
            int idx = t + i*256;            // 512 chunks of 16B
            int row = idx >> 2;
            int ch  = (idx & 3) << 3;       // element offset
            if (n0 + row >= Nout) {
                const float4 z = make_float4(0.f,0.f,0.f,0.f);
                *(float4*)(void*)&Bs[0][row*PK + ch] = z;
                *(float4*)(void*)&Bs[1][row*PK + ch] = z;
            }
        }
    }

    auto stage = [&](int kt, int buf) {
        const int k0 = kt * BK;
#pragma unroll
        for (int i = 0; i < 2; ++i) {       // A tile: 512 x 16B chunks
            int idx = t + i*256;
            int row = idx >> 2;
            int ch  = (idx & 3) << 3;
            async_cp16((lds_bf16*)&As[buf][row*PK + ch],
                       A + (size_t)(m0+row)*lda + k0 + ch);
        }
#pragma unroll
        for (int i = 0; i < 2; ++i) {       // B tile: 512 x 16B chunks
            int idx = t + i*256;
            int row = idx >> 2;
            int ch  = (idx & 3) << 3;
            const bool ok = (n0 + row < Nout);
            lds_bf16* dst = ok ? (lds_bf16*)&Bs[buf][row*PK + ch]
                               : (lds_bf16*)&dump[0];
            const int srow = ok ? (n0 + row) : (Nout - 1);
            async_cp16(dst, W + (size_t)srow*Kdim + k0 + ch);
        }
    };

    const v8f zero = {};
    v8f acc[4][2];
#pragma unroll
    for (int i = 0; i < 4; ++i)
#pragma unroll
        for (int j = 0; j < 2; ++j) acc[i][j] = zero;

    auto compute = [&](int cur) {
        v16bf af[4], bfr[2];
#pragma unroll
        for (int i = 0; i < 4; ++i) {
            const __bf16* p = &As[cur][(wm*64 + i*16 + lr)*PK + lh*8];
            v8bf lo = *(const v8bf*)p;
            v8bf hi = *(const v8bf*)(p + 16);
            af[i] = __builtin_shufflevector(lo, hi, 0,1,2,3,4,5,6,7,8,9,10,11,12,13,14,15);
        }
#pragma unroll
        for (int j = 0; j < 2; ++j) {
            const __bf16* p = &Bs[cur][(wn*32 + j*16 + lr)*PK + lh*8];
            v8bf lo = *(const v8bf*)p;
            v8bf hi = *(const v8bf*)(p + 16);
            bfr[j] = __builtin_shufflevector(lo, hi, 0,1,2,3,4,5,6,7,8,9,10,11,12,13,14,15);
        }
#pragma unroll
        for (int i = 0; i < 4; ++i)
#pragma unroll
            for (int j = 0; j < 2; ++j)
                acc[i][j] = __builtin_amdgcn_wmma_f32_16x16x32_bf16(
                    false, af[i], false, bfr[j], (short)0, acc[i][j], false, false);
    };

    const int KT = Kdim / BK;
    stage(0, 0);
    // steady state: prefetch next tile, wait for current, compute
    for (int kt = 0; kt < KT - 1; ++kt) {
        const int cur = kt & 1;
        stage(kt + 1, cur ^ 1);
        wait_asynccnt<4>();
        __syncthreads();
        compute(cur);
        __syncthreads();
    }
    // last tile
    wait_asynccnt<0>();
    __syncthreads();
    compute((KT - 1) & 1);

    // epilogue: C/D layout -> row = base + r + 8*lh, col = base + lr
#pragma unroll
    for (int i = 0; i < 4; ++i) {
#pragma unroll
        for (int j = 0; j < 2; ++j) {
            const int col = n0 + wn*32 + j*16 + lr;
            if (col >= Nout) continue;
            const float bb = bias ? bias[col] : 0.f;
#pragma unroll
            for (int r = 0; r < 8; ++r) {
                const int row = m0 + wm*64 + i*16 + lh*8 + r;
                float v = acc[i][j][r] + bb;
                if (actmode == 1) v = softplusf(v);
                if (resid) v += resid[(size_t)row*ldc + col];
                C[(size_t)row*ldc + col] = v;
            }
        }
    }
}

// ----------------------------------------------------------------------------
// GLU-fused bf16 WMMA GEMM for fc1: act[M,FF] = bf16( yy * silu(g) )
//   g = A*W[0:FF]^T, yy = A*W[FF:2FF]^T; BM=128 BN=64; two accumulator sets.
// ----------------------------------------------------------------------------
__global__ __launch_bounds__(256)
void gemm_glu_wmma(const __bf16* __restrict__ A,   // [M, DM]
                   const __bf16* __restrict__ W,   // [2FF, DM]
                   __bf16* __restrict__ OUT)       // [M, FF]
{
    constexpr int BM = 128, BN = 64, BK = 32, PK = BK + 8, KD = DMf;
    __shared__ __align__(16) __bf16 As [2][BM*PK];
    __shared__ __align__(16) __bf16 Bs0[2][BN*PK];
    __shared__ __align__(16) __bf16 Bs1[2][BN*PK];

    const int t    = threadIdx.x;
    const int lane = t & 31, wave = t >> 5;
    const int wm   = wave >> 1, wn = wave & 1;      // 4 x 2 wave grid
    const int lr   = lane & 15, lh = lane >> 4;
    const int m0   = blockIdx.y * BM;
    const int n0   = blockIdx.x * BN;

    auto stage = [&](int kt, int buf) {
        const int k0 = kt * BK;
#pragma unroll
        for (int i = 0; i < 2; ++i) {       // A tile: 512 x 16B chunks
            int idx = t + i*256;
            int row = idx >> 2;
            int ch  = (idx & 3) << 3;
            async_cp16((lds_bf16*)&As[buf][row*PK + ch],
                       A + (size_t)(m0+row)*KD + k0 + ch);
        }
        {                                   // B0/B1: 256 x 16B chunks each
            int row = t >> 2;
            int ch  = (t & 3) << 3;
            async_cp16((lds_bf16*)&Bs0[buf][row*PK + ch],
                       W + (size_t)(n0+row)*KD + k0 + ch);
            async_cp16((lds_bf16*)&Bs1[buf][row*PK + ch],
                       W + (size_t)(FFDf + n0 + row)*KD + k0 + ch);
        }
    };

    const v8f zero = {};
    v8f cg[2][2], cy[2][2];
#pragma unroll
    for (int i = 0; i < 2; ++i)
#pragma unroll
        for (int j = 0; j < 2; ++j) { cg[i][j] = zero; cy[i][j] = zero; }

    auto compute = [&](int cur) {
        v16bf af[2], b0[2], b1[2];
#pragma unroll
        for (int i = 0; i < 2; ++i) {
            const __bf16* p = &As[cur][(wm*32 + i*16 + lr)*PK + lh*8];
            v8bf lo = *(const v8bf*)p;
            v8bf hi = *(const v8bf*)(p + 16);
            af[i] = __builtin_shufflevector(lo, hi, 0,1,2,3,4,5,6,7,8,9,10,11,12,13,14,15);
        }
#pragma unroll
        for (int j = 0; j < 2; ++j) {
            const __bf16* p = &Bs0[cur][(wn*32 + j*16 + lr)*PK + lh*8];
            v8bf lo = *(const v8bf*)p;
            v8bf hi = *(const v8bf*)(p + 16);
            b0[j] = __builtin_shufflevector(lo, hi, 0,1,2,3,4,5,6,7,8,9,10,11,12,13,14,15);
            const __bf16* q = &Bs1[cur][(wn*32 + j*16 + lr)*PK + lh*8];
            v8bf lo1 = *(const v8bf*)q;
            v8bf hi1 = *(const v8bf*)(q + 16);
            b1[j] = __builtin_shufflevector(lo1, hi1, 0,1,2,3,4,5,6,7,8,9,10,11,12,13,14,15);
        }
#pragma unroll
        for (int i = 0; i < 2; ++i)
#pragma unroll
            for (int j = 0; j < 2; ++j) {
                cg[i][j] = __builtin_amdgcn_wmma_f32_16x16x32_bf16(
                    false, af[i], false, b0[j], (short)0, cg[i][j], false, false);
                cy[i][j] = __builtin_amdgcn_wmma_f32_16x16x32_bf16(
                    false, af[i], false, b1[j], (short)0, cy[i][j], false, false);
            }
    };

    const int KT = KD / BK;
    stage(0, 0);
    for (int kt = 0; kt < KT - 1; ++kt) {
        const int cur = kt & 1;
        stage(kt + 1, cur ^ 1);
        wait_asynccnt<4>();
        __syncthreads();
        compute(cur);
        __syncthreads();
    }
    wait_asynccnt<0>();
    __syncthreads();
    compute((KT - 1) & 1);

#pragma unroll
    for (int i = 0; i < 2; ++i) {
#pragma unroll
        for (int j = 0; j < 2; ++j) {
            const int col = n0 + wn*32 + j*16 + lr;
#pragma unroll
            for (int r = 0; r < 8; ++r) {
                const int row = m0 + wm*32 + i*16 + lh*8 + r;
                OUT[(size_t)row*FFDf + col] = (__bf16)(cy[i][j][r] * siluf(cg[i][j][r]));
            }
        }
    }
}

// ----------------------------------------------------------------------------
// Sequential SSM scan over L, fused D-skip + SiLU(gate); bf16 output.
// One thread per (channel d, state n); 16-lane shfl_xor reduction for h.C.
// ----------------------------------------------------------------------------
__global__ __launch_bounds__(256)
void ssm_scan_kernel(const float* __restrict__ ssm,    // [M,160]  (dt | B | C)
                     const float* __restrict__ delta,  // [M,DI]
                     const float* __restrict__ xs,     // [M,DI]
                     const float* __restrict__ xz,     // [M,2DI] (gate at DI+d)
                     const float* __restrict__ Am,     // [DI,16]
                     const float* __restrict__ Dp,     // [DI]
                     __bf16* __restrict__ yb)          // [M,DI] bf16
{
    const int t    = threadIdx.x;
    const int n    = t & 15;
    const int chan = blockIdx.x * 16 + (t >> 4);   // 0 .. B*DI-1
    const int b    = chan / DIf;
    const int d    = chan % DIf;

    const float a  = Am[d*NSTf + n];
    const float dp = Dp[d];
    float h = 0.f;

    for (int l = 0; l < LLf; ++l) {
        const size_t bl  = (size_t)b*LLf + l;
        const size_t idx = bl*DIf + d;
        const float dlt = delta[idx];
        const float xt  = xs[idx];
        const float Bn  = ssm[bl*160 + RRf + n];
        const float Cn  = ssm[bl*160 + RRf + NSTf + n];

        h = __expf(dlt * a) * h + (dlt * xt) * Bn;
        float p = h * Cn;
        p += __shfl_xor(p, 8, 16);
        p += __shfl_xor(p, 4, 16);
        p += __shfl_xor(p, 2, 16);
        p += __shfl_xor(p, 1, 16);
        if (n == 0) {
            const float g = xz[bl*(size_t)(2*DIf) + DIf + d];
            yb[idx] = (__bf16)((p + dp * xt) * siluf(g));
        }
    }
}

// ----------------------------------------------------------------------------
// Launch
// ----------------------------------------------------------------------------
extern "C" void kernel_launch(void* const* d_in, const int* in_sizes, int n_in,
                              void* d_out, int out_size, void* d_ws, size_t ws_size,
                              hipStream_t stream)
{
    (void)in_sizes; (void)n_in; (void)out_size; (void)ws_size;

    const float* hidden = (const float*)d_in[0];
    const float* ln1w   = (const float*)d_in[1];
    const float* ln1b   = (const float*)d_in[2];
    const float* W_in   = (const float*)d_in[3];
    const float* conv_w = (const float*)d_in[4];
    const float* conv_b = (const float*)d_in[5];
    const float* W_x    = (const float*)d_in[6];
    const float* W_dt   = (const float*)d_in[7];
    const float* b_dt   = (const float*)d_in[8];
    const float* Am     = (const float*)d_in[9];
    const float* Dp     = (const float*)d_in[10];
    const float* W_out  = (const float*)d_in[11];
    const float* ln2w   = (const float*)d_in[12];
    const float* ln2b   = (const float*)d_in[13];
    const float* W_fc1  = (const float*)d_in[14];
    const float* W_fc2  = (const float*)d_in[15];
    float* out = (float*)d_out;

    char* ws = (char*)d_ws;
    size_t off = 0;
    auto walloc = [&](size_t bytes) -> void* {
        void* p = ws + off;
        off = (off + bytes + 255) & ~(size_t)255;
        return p;
    };
    // bf16 weight copies (cast once per call)
    __bf16* w_in_bf  = (__bf16*)walloc((size_t)2*DIf*DMf*2);     // 32 MB
    __bf16* w_x_bf   = (__bf16*)walloc((size_t)160*DIf*2);
    __bf16* w_dt_bf  = (__bf16*)walloc((size_t)DIf*RRf*2);
    __bf16* w_out_bf = (__bf16*)walloc((size_t)DMf*DIf*2);       // 16 MB
    __bf16* w_fc1_bf = (__bf16*)walloc((size_t)2*FFDf*DMf*2);    // 64 MB
    __bf16* w_fc2_bf = (__bf16*)walloc((size_t)DMf*FFDf*2);      // 32 MB
    // activations
    __bf16* x_bf   = (__bf16*)walloc((size_t)MROWS*DMf*2);       // 16 MB LN out (reused for LN2)
    float*  xz     = (float* )walloc((size_t)MROWS*2*DIf*4);     // 128 MB
    float*  xs     = (float* )walloc((size_t)MROWS*DIf*4);       // 64 MB
    __bf16* xs_bf  = (__bf16*)walloc((size_t)MROWS*DIf*2);       // 32 MB
    float*  ssm    = (float* )walloc((size_t)MROWS*160*4);       // 2.6 MB
    __bf16* dt_bf  = (__bf16*)walloc((size_t)MROWS*RRf*2);       // 1 MB
    float*  delta  = (float* )walloc((size_t)MROWS*DIf*4);       // 64 MB
    __bf16* y_bf   = (__bf16*)walloc((size_t)MROWS*DIf*2);       // 32 MB
    float*  hs     = (float* )walloc((size_t)MROWS*DMf*4);       // 32 MB
    __bf16* x2_bf  = x_bf;                                       // reuse
    __bf16* act_bf = (__bf16*)xz;                                // reuse (xz dead after scan)

    const dim3 blk(256);

    // 0) cast weights to bf16
    cast_bf16_kernel<<<(unsigned)((size_t)2*DIf*DMf/2048),  blk, 0, stream>>>(W_in,  w_in_bf);
    cast_bf16_kernel<<<(unsigned)((size_t)160*DIf/2048),    blk, 0, stream>>>(W_x,   w_x_bf);
    cast_bf16_kernel<<<(unsigned)((size_t)DIf*RRf/2048),    blk, 0, stream>>>(W_dt,  w_dt_bf);
    cast_bf16_kernel<<<(unsigned)((size_t)DMf*DIf/2048),    blk, 0, stream>>>(W_out, w_out_bf);
    cast_bf16_kernel<<<(unsigned)((size_t)2*FFDf*DMf/2048), blk, 0, stream>>>(W_fc1, w_fc1_bf);
    cast_bf16_kernel<<<(unsigned)((size_t)DMf*FFDf/2048),   blk, 0, stream>>>(W_fc2, w_fc2_bf);

    // 1) LN1 -> bf16
    ln_kernel<<<MROWS, blk, 0, stream>>>(hidden, ln1w, ln1b, x_bf);
    // 2) xz = x @ W_in^T                    [4096 x 8192 x 2048]
    gemm_wmma_bf16<<<dim3(2*DIf/128, MROWS/128), blk, 0, stream>>>(
        x_bf, DMf, w_in_bf, 2*DIf, DMf, xz, 2*DIf, nullptr, nullptr, 0);
    // 3) causal conv + SiLU -> xs (f32 + bf16)
    conv_silu_kernel<<<(unsigned)((size_t)MROWS*DIf/256), blk, 0, stream>>>(
        xz, conv_w, conv_b, xs, xs_bf);
    // 4) ssm = xs @ W_x^T                   [4096 x 160 x 4096]
    gemm_wmma_bf16<<<dim3(2, MROWS/128), blk, 0, stream>>>(
        xs_bf, DIf, w_x_bf, 160, DIf, ssm, 160, nullptr, nullptr, 0);
    // 5) dt cast + delta = softplus(dt @ W_dt^T + b_dt)  [4096 x 4096 x 128]
    cast_dt_kernel<<<(unsigned)((size_t)MROWS*RRf/1024), blk, 0, stream>>>(ssm, dt_bf);
    gemm_wmma_bf16<<<dim3(DIf/128, MROWS/128), blk, 0, stream>>>(
        dt_bf, RRf, w_dt_bf, DIf, RRf, delta, DIf, b_dt, nullptr, 1);
    // 6) sequential scan, fused D-skip + SiLU(gate) -> y (bf16)
    ssm_scan_kernel<<<BBf*DIf/16, blk, 0, stream>>>(ssm, delta, xs, xz, Am, Dp, y_bf);
    // 7) hs = y @ W_out^T + hidden          [4096 x 2048 x 4096]
    gemm_wmma_bf16<<<dim3(DMf/128, MROWS/128), blk, 0, stream>>>(
        y_bf, DIf, w_out_bf, DMf, DIf, hs, DMf, nullptr, hidden, 0);
    // 8) LN2 -> bf16
    ln_kernel<<<MROWS, blk, 0, stream>>>(hs, ln2w, ln2b, x2_bf);
    // 9) act = bf16(silu(g) * yy)  (GLU-fused fc1)
    gemm_glu_wmma<<<dim3(FFDf/64, MROWS/128), blk, 0, stream>>>(x2_bf, w_fc1_bf, act_bf);
    // 10) out = act @ W_fc2^T + hs          [4096 x 2048 x 8192]
    gemm_wmma_bf16<<<dim3(DMf/128, MROWS/128), blk, 0, stream>>>(
        act_bf, FFDf, w_fc2_bf, DMf, FFDf, out, DMf, nullptr, hs, 0);
}
